// _ProposalLayer_55353538511020
// MI455X (gfx1250) — matrix operand order repaired and checked by reading, not compiled, not run on hardware
//
#include <hip/hip_runtime.h>
#include <hip/hip_bf16.h>
#include <math.h>

typedef __attribute__((ext_vector_type(16))) _Float16     v16h;
typedef __attribute__((ext_vector_type(8)))  float        v8f;
typedef __attribute__((ext_vector_type(4)))  unsigned int u32x4;
typedef __attribute__((ext_vector_type(8)))  int          i32x8;
typedef __attribute__((ext_vector_type(4)))  int          i32x4;

#define B_      2
#define A_      9
#define H_      50
#define W_      50
#define HW_     (H_ * W_)          // 2500
#define N_      (A_ * HW_)         // 22500
#define PRE_    4000
#define POST_   300
#define THRESH_ 0.7f
#define STRIDE_ 16.0f

// workspace layout (float offsets)
#define OFF_PROPS   0                          // B*N*4  = 180000
#define OFF_SCORES  (OFF_PROPS  + B_*N_*4)     // B*N    =  45000
#define OFF_SORTED  (OFF_SCORES + B_*N_)       // B*PRE*4=  32000
#define OFF_KEEP    (OFF_SORTED + B_*PRE_*4)   // B*4096 =   8192

// ---------------------------------------------------------------------------
// Tensor Data Mover: DMA `nfloats` contiguous f32 global -> LDS.
// D# per CDNA5 ISA 8.3/8.4: group0 = {count/type, lds_addr, global_addr},
// group1 = {data_size=4B, tensor_dim0=tile_dim0=nfloats, 1 row}.
// Caller must be wave-uniform; completion via s_wait_tensorcnt.
// ---------------------------------------------------------------------------
__device__ static inline void tdm_load_to_lds(const float* gsrc, void* lds_dst,
                                              unsigned int nfloats)
{
    unsigned long long ga  = (unsigned long long)(const void*)gsrc;
    unsigned int       lds = (unsigned int)(unsigned long long)lds_dst; // LDS offset = flat[31:0]

    u32x4 g0;
    g0[0] = 1u;                                         // count=1 (valid), user desc
    g0[1] = lds;                                        // lds_addr [63:32]
    g0[2] = (unsigned int)(ga & 0xFFFFFFFFull);         // global_addr[31:0]
    g0[3] = (unsigned int)((ga >> 32) & 0x01FFFFFFull)  // global_addr[56:32]
          | 0x80000000u;                                // type=2 ("image")

    i32x8 g1;
    g1[0] = (int)(2u << 16);                            // wg_mask=0 | data_size=2 (4B)
    g1[1] = (int)((nfloats & 0xFFFFu) << 16);           // tensor_dim0[15:0] @ bits 63:48
    g1[2] = (int)(((nfloats >> 16) & 0xFFFFu)           // tensor_dim0[31:16]
          | (1u << 16));                                // tensor_dim1 = 1
    g1[3] = (int)((nfloats & 0xFFFFu) << 16);           // tile_dim0 @ bits 127:112
    g1[4] = 1;                                          // tile_dim1=1, tile_dim2=0
    g1[5] = (int)nfloats;                               // tensor_dim0_stride[31:0]
    g1[6] = (int)((nfloats & 0xFFFFu) << 16);           // stride0[47:32]=0 | stride1[15:0]
    g1[7] = 0;                                          // stride1[47:16]=0

    i32x4 gz = {0, 0, 0, 0};
#if defined(__clang_major__) && (__clang_major__ >= 23)
    i32x8 gz8 = {0, 0, 0, 0, 0, 0, 0, 0};
    __builtin_amdgcn_tensor_load_to_lds(g0, g1, gz, gz, gz8, 0);
#else
    __builtin_amdgcn_tensor_load_to_lds(g0, g1, gz, gz, 0);
#endif
}

// ---------------------------------------------------------------------------
// K1: decode proposals (anchor shift + delta transform + clip) and stash
//     per-proposal score; also initialize the 600 output rows to [b,0,0,0,0].
// ---------------------------------------------------------------------------
__global__ void k_decode(const float* __restrict__ scores,
                         const float* __restrict__ deltas,
                         const float* __restrict__ iminfo,
                         const float* __restrict__ anchors,
                         float* __restrict__ props,
                         float* __restrict__ scbuf,
                         float* __restrict__ out)
{
    int i = blockIdx.x * blockDim.x + threadIdx.x;

    if (i < B_ * POST_) {
        int b = i / POST_;
        float* row = out + (size_t)i * 5;
        row[0] = (float)b;
        row[1] = 0.0f; row[2] = 0.0f; row[3] = 0.0f; row[4] = 0.0f;
    }
    if (i >= B_ * N_) return;

    int b  = i / N_;
    int n  = i - b * N_;
    int a  = n % A_;             // proposal index n = (h*W + w)*A + a
    int hw = n / A_;
    int h  = hw / W_;
    int w  = hw - h * W_;

    float sx = (float)w * STRIDE_;
    float sy = (float)h * STRIDE_;
    float ax1 = anchors[a * 4 + 0] + sx;
    float ay1 = anchors[a * 4 + 1] + sy;
    float ax2 = anchors[a * 4 + 2] + sx;
    float ay2 = anchors[a * 4 + 3] + sy;

    float wid = ax2 - ax1 + 1.0f;
    float hei = ay2 - ay1 + 1.0f;
    float cx  = ax1 + 0.5f * wid;
    float cy  = ay1 + 0.5f * hei;

    const float* dbase = deltas + ((size_t)b * 36 + 4 * a) * HW_ + hw;
    float dx = dbase[0];
    float dy = dbase[HW_];
    float dw = dbase[2 * HW_];
    float dh = dbase[3 * HW_];

    float pcx = dx * wid + cx;
    float pcy = dy * hei + cy;
    float pw  = expf(dw) * wid;
    float ph  = expf(dh) * hei;

    float px1 = pcx - 0.5f * pw;
    float py1 = pcy - 0.5f * ph;
    float px2 = pcx + 0.5f * pw;
    float py2 = pcy + 0.5f * ph;

    float wmax = iminfo[b * 3 + 1] - 1.0f;
    float hmax = iminfo[b * 3 + 0] - 1.0f;
    px1 = fminf(fmaxf(px1, 0.0f), wmax);
    px2 = fminf(fmaxf(px2, 0.0f), wmax);
    py1 = fminf(fmaxf(py1, 0.0f), hmax);
    py2 = fminf(fmaxf(py2, 0.0f), hmax);

    float* p = props + ((size_t)b * N_ + n) * 4;
    p[0] = px1; p[1] = py1; p[2] = px2; p[3] = py2;

    // objectness channels are A..2A-1
    scbuf[(size_t)b * N_ + n] = scores[((size_t)b * 18 + 9 + a) * HW_ + hw];
}

// ---------------------------------------------------------------------------
// K2: exact stable top-PRE_ selection by rank counting (descending score,
//     ties -> lower index, matching jax.lax.top_k). LDS-tiled scan with
//     global_prefetch one tile ahead (gfx1250 prefetch path).
// ---------------------------------------------------------------------------
__global__ void k_select(const float* __restrict__ scbuf,
                         const float* __restrict__ props,
                         float* __restrict__ sorted)
{
    __shared__ float tile[256];
    int b = blockIdx.y;
    int n = blockIdx.x * 256 + threadIdx.x;
    const float* sc = scbuf + (size_t)b * N_;

    float s = (n < N_) ? sc[n] : -3.0e38f;
    int rank = 0;

    for (int base = 0; base < N_; base += 256) {
        int m = base + threadIdx.x;
        tile[threadIdx.x] = (m < N_) ? sc[m] : -3.0e38f;
        if (base + 256 + (int)threadIdx.x < N_)
            __builtin_prefetch(&sc[base + 256 + threadIdx.x], 0, 0);
        __syncthreads();
        int lim = N_ - base; if (lim > 256) lim = 256;
        for (int t = 0; t < lim; ++t) {
            float ms = tile[t];
            int   mi = base + t;
            rank += ((ms > s) || (ms == s && mi < n)) ? 1 : 0;
        }
        __syncthreads();
    }

    if (n < N_ && rank < PRE_) {
        const float* p = props  + ((size_t)b * N_   + n)    * 4;
        float*       q = sorted + ((size_t)b * PRE_ + rank) * 4;
        q[0] = p[0]; q[1] = p[1]; q[2] = p[2]; q[3] = p[3];
    }
}

// ---------------------------------------------------------------------------
// K3: greedy NMS, one workgroup per image. Boxes staged into LDS by the
//     Tensor Data Mover (64000 B flat DMA, TENSORcnt-tracked), issued by
//     wave 0 only. Writes keep flags (f32 0/1) padded to 4096 per image.
// ---------------------------------------------------------------------------
__global__ void k_nms(const float* __restrict__ sorted,
                      float* __restrict__ keepf)
{
    __shared__ float4 box[PRE_];               // 64000 B, x1,y1,x2,y2 interleaved
    __shared__ unsigned int sup[PRE_ / 32];    // 500 B

    int b = blockIdx.x;
    const float* base = sorted + (size_t)b * PRE_ * 4;

    if (threadIdx.x < 32) {                    // wave-uniform: only wave 0 issues DMA
        tdm_load_to_lds(base, (void*)&box[0], PRE_ * 4);
        __builtin_amdgcn_s_wait_tensorcnt(0);
    }
    for (int j = threadIdx.x; j < PRE_ / 32; j += blockDim.x) sup[j] = 0u;
    __syncthreads();

    for (int i = 0; i < PRE_; ++i) {
        // bit i is final once iteration i starts (only bits > i mutate now)
        unsigned supi = (sup[i >> 5] >> (i & 31)) & 1u;
        if (!supi) {
            float4 bi = box[i];
            float iarea = (bi.z - bi.x + 1.0f) * (bi.w - bi.y + 1.0f);
            for (int j = i + 1 + (int)threadIdx.x; j < PRE_; j += blockDim.x) {
                float4 bj = box[j];
                float xx1 = fmaxf(bi.x, bj.x);
                float yy1 = fmaxf(bi.y, bj.y);
                float xx2 = fminf(bi.z, bj.z);
                float yy2 = fminf(bi.w, bj.w);
                float ww = fmaxf(xx2 - xx1 + 1.0f, 0.0f);
                float hh = fmaxf(yy2 - yy1 + 1.0f, 0.0f);
                float inter = ww * hh;
                float jarea = (bj.z - bj.x + 1.0f) * (bj.w - bj.y + 1.0f);
                float iou = inter / (iarea + jarea - inter);
                if (iou > THRESH_) atomicOr(&sup[j >> 5], 1u << (j & 31));
            }
        }
        __syncthreads();
    }

    for (int j = threadIdx.x; j < 4096; j += blockDim.x) {
        float k = 0.0f;
        if (j < PRE_) k = ((sup[j >> 5] >> (j & 31)) & 1u) ? 0.0f : 1.0f;
        keepf[(size_t)b * 4096 + j] = k;
    }
}

// ---------------------------------------------------------------------------
// K4: rank = cumsum(keep)-1 computed on the matrix unit.
//     Prefix-sum == lower-triangular-ones matmul:
//       A (16x32 f16) = [L | 0], L[r][k] = (k<=r)
//       B (32x16 f16) = 16 blocks of 16 keep flags (rows 16..31 zero)
//       D (16x16 f32) = intra-block inclusive prefixes for 256 flags/WMMA.
//     16 WMMAs cover 4096 flags; block carries are exact small ints.
//     One wave (32 lanes, EXEC all ones) per image.
// ---------------------------------------------------------------------------
__global__ void k_rank_out(const float* __restrict__ sorted,
                           const float* __restrict__ keepf,
                           float* __restrict__ out)
{
    __shared__ int ltot[16];
    int b = blockIdx.x;
    int l = threadIdx.x;                     // 0..31
    const float* kf = keepf + (size_t)b * 4096;

    // A-matrix in contiguous K order (B uses the same): lane=M(mod 16); elem e -> K in
    // {0..15} (lanes 0-15) / {16..31} (lanes 16-31).
    v16h amat;
#pragma unroll
    for (int e = 0; e < 16; ++e) {
        int K = ((l < 16) ? 0 : 16) + e;
        int M = l & 15;
        amat[e] = (_Float16)((K <= M) ? 1.0f : 0.0f);
    }

    int carry = 0;
    int Nc = l & 15;                         // this lane's block column
    float msk = (l < 16) ? 1.0f : 0.0f;      // lanes 16..31 supply K=16..31 zeros
    for (int it = 0; it < 16; ++it) {
        int blk0 = it * 16;                  // first block of this group

        // B-matrix in the same contiguous K order: lanes 0-15 hold K=e (rows 0..15),
        // lanes 16-31 hold K=16+e (zero rows), N = lane%16. Vectorized: the
        // 16 flags of block N are contiguous -> 4x global_load_b128 + mask.
        const float4* kp4 = (const float4*)(kf + (blk0 + Nc) * 16);
        float4 q0 = kp4[0], q1 = kp4[1], q2 = kp4[2], q3 = kp4[3];
        v16h bmat;
        bmat[0]  = (_Float16)(q0.x * msk); bmat[1]  = (_Float16)(q0.y * msk);
        bmat[2]  = (_Float16)(q0.z * msk); bmat[3]  = (_Float16)(q0.w * msk);
        bmat[4]  = (_Float16)(q1.x * msk); bmat[5]  = (_Float16)(q1.y * msk);
        bmat[6]  = (_Float16)(q1.z * msk); bmat[7]  = (_Float16)(q1.w * msk);
        bmat[8]  = (_Float16)(q2.x * msk); bmat[9]  = (_Float16)(q2.y * msk);
        bmat[10] = (_Float16)(q2.z * msk); bmat[11] = (_Float16)(q2.w * msk);
        bmat[12] = (_Float16)(q3.x * msk); bmat[13] = (_Float16)(q3.y * msk);
        bmat[14] = (_Float16)(q3.z * msk); bmat[15] = (_Float16)(q3.w * msk);

        v8f acc = {};
        acc = __builtin_amdgcn_wmma_f32_16x16x32_f16(
                  false, amat, false, bmat, (short)0, acc, false, false);

        // D elem e in lane l -> M = e + (l<16 ? 0 : 8), N = l%16.
        // Block total = D[15][N] = elem 7 of lanes 16..31.
        if (l >= 16) ltot[l - 16] = (int)(acc[7] + 0.5f);
        __syncthreads();

        int excl = 0, tot = 0;
        for (int c = 0; c < 16; ++c) {
            int t = ltot[c];
            if (c < Nc) excl += t;
            tot += t;
        }

#pragma unroll
        for (int e = 0; e < 8; ++e) {
            int pos = e + ((l < 16) ? 0 : 8);
            int idx = (blk0 + Nc) * 16 + pos;
            if (idx < PRE_) {
                bool kept = kf[idx] > 0.5f;
                int rank = carry + excl + (int)(acc[e] + 0.5f) - 1;
                if (kept && rank < POST_) {
                    const float* p = sorted + ((size_t)b * PRE_ + idx) * 4;
                    float* row = out + ((size_t)b * POST_ + rank) * 5;
                    row[0] = (float)b;
                    row[1] = p[0]; row[2] = p[1]; row[3] = p[2]; row[4] = p[3];
                }
            }
        }
        carry += tot;
        __syncthreads();
    }
}

// ---------------------------------------------------------------------------
extern "C" void kernel_launch(void* const* d_in, const int* in_sizes, int n_in,
                              void* d_out, int out_size, void* d_ws, size_t ws_size,
                              hipStream_t stream) {
    const float* scores  = (const float*)d_in[0];   // (2,18,50,50)
    const float* deltas  = (const float*)d_in[1];   // (2,36,50,50)
    const float* iminfo  = (const float*)d_in[2];   // (2,3)
    const float* anchors = (const float*)d_in[3];   // (9,4)
    float* out = (float*)d_out;                     // (2,300,5)
    float* ws  = (float*)d_ws;

    float* props  = ws + OFF_PROPS;
    float* scbuf  = ws + OFF_SCORES;
    float* sorted = ws + OFF_SORTED;
    float* keepf  = ws + OFF_KEEP;

    int total = B_ * N_;
    k_decode<<<(total + 255) / 256, 256, 0, stream>>>(scores, deltas, iminfo,
                                                      anchors, props, scbuf, out);

    dim3 g2((N_ + 255) / 256, B_);
    k_select<<<g2, 256, 0, stream>>>(scbuf, props, sorted);

    k_nms<<<B_, 256, 0, stream>>>(sorted, keepf);

    k_rank_out<<<B_, 32, 0, stream>>>(sorted, keepf, out);
}
